// motionNet_26912265076922
// MI455X (gfx1250) — matrix operand order repaired and checked by reading, not compiled, not run on hardware
//
#include <hip/hip_runtime.h>
#include <hip/hip_bf16.h>
#include <stdint.h>

// Problem dims (compile-time constants from the reference)
#define T_DIM 1024
#define B_DIM 128
#define I_DIM 512
#define H_DIM 512
#define O_DIM 4
#define ALPHA_F 0.083f
#define NWG_SCAN 16

typedef __attribute__((ext_vector_type(16))) __bf16 bf16x16;
typedef __attribute__((ext_vector_type(8)))  __bf16 bf16x8;
typedef __attribute__((ext_vector_type(8)))  float  f32x8;
typedef unsigned int u32x4 __attribute__((ext_vector_type(4)));
typedef int          i32x4 __attribute__((ext_vector_type(4)));
typedef int          i32x8 __attribute__((ext_vector_type(8)));

static __device__ __forceinline__ bf16x16 combine16(bf16x8 lo, bf16x8 hi) {
    bf16x16 r;
#pragma unroll
    for (int i = 0; i < 8; ++i) { r[i] = lo[i]; r[i + 8] = hi[i]; }
    return r;
}

// A-matrix fragment (16x32 bf16, row-major source, stride ld):
// lanes 0-15: row=lane,   K chunks [0..7] and [16..23]
// lanes 16-31: row=lane-16, K chunks [8..15] and [24..31]
static __device__ __forceinline__ bf16x16 load_a_bf16(const __bf16* base, int ld, int lane) {
    int r = lane & 15, half = lane >> 4;
    const __bf16* p = base + (size_t)r * ld + half * 8;
    bf16x8 c0 = *(const bf16x8*)(p);
    bf16x8 c1 = *(const bf16x8*)(p + 16);
    return combine16(c0, c1);
}

// Same A layout but converting from f32 source on the fly
static __device__ __forceinline__ bf16x16 load_a_f32cvt(const float* base, int ld, int lane) {
    int r = lane & 15, half = lane >> 4;
    const float* p = base + (size_t)r * ld + half * 8;
    f32x8 c0 = *(const f32x8*)(p);
    f32x8 c1 = *(const f32x8*)(p + 16);
    bf16x16 a;
#pragma unroll
    for (int i = 0; i < 8; ++i) { a[i] = (__bf16)c0[i]; a[i + 8] = (__bf16)c1[i]; }
    return a;
}

// B-matrix fragment (32x16 = KxN). Source is W[n][k] row-major (stride ld),
// so column n of B == row n of W. lane holds column (lane&15);
// lanes 0-15 -> K 0..7,16..23; lanes 16-31 -> K 8..15,24..31 (same as A).
static __device__ __forceinline__ bf16x16 load_b_bf16(const __bf16* base, int ld, int lane) {
    int c = lane & 15, half = lane >> 4;
    const __bf16* p = base + (size_t)c * ld + half * 8;
    bf16x8 c0 = *(const bf16x8*)(p);
    bf16x8 c1 = *(const bf16x8*)(p + 16);
    return combine16(c0, c1);
}

// ---------------------------------------------------------------------------
// Prep: convert weights f32 -> bf16 (both 512x512 matrices in one pass)
__global__ __launch_bounds__(256) void rnn_cvt_weights_k(
    const float* __restrict__ Win, const float* __restrict__ Wh,
    __bf16* __restrict__ Winb, __bf16* __restrict__ Whb) {
    int i = blockIdx.x * 256 + threadIdx.x;
    if (i < H_DIM * I_DIM) {
        Winb[i] = (__bf16)Win[i];
        Whb[i]  = (__bf16)Wh[i];
    }
}

// Init: zero h state (f32 + both bf16 buffers) and the grid-barrier counter
__global__ __launch_bounds__(256) void rnn_init_k(
    float* __restrict__ h_f32, __bf16* __restrict__ hb0, __bf16* __restrict__ hb1,
    unsigned* __restrict__ cnt) {
    int i = blockIdx.x * 256 + threadIdx.x;
    if (i < B_DIM * H_DIM) {
        h_f32[i] = 0.0f;
        hb0[i] = (__bf16)0.0f;
        hb1[i] = (__bf16)0.0f;
    }
    if (i == 0) *cnt = 0u;
}

// ---------------------------------------------------------------------------
// Phase 1: xproj[m, n] = sum_k x[m,k] * W_in[n,k] + b_in[n], stored bf16.
// WG = 256 threads = 8 waves arranged 4(M) x 2(N); WG tile 128x64.
// Wave tile 32x32 = 2x2 WMMA accumulators; K-loop over 512 in steps of 32.
__global__ __launch_bounds__(256) void rnn_xproj_gemm_k(
    const float* __restrict__ x, const __bf16* __restrict__ Winb,
    const float* __restrict__ b_in, __bf16* __restrict__ xproj) {
    int lane = threadIdx.x & 31;
    int w = threadIdx.x >> 5;
    int wm = w >> 1, wn = w & 1;
    int Mbase = blockIdx.y * 128 + wm * 32;
    int Nbase = blockIdx.x * 64 + wn * 32;

    f32x8 acc[2][2];
#pragma unroll
    for (int i = 0; i < 2; ++i)
#pragma unroll
        for (int j = 0; j < 2; ++j) acc[i][j] = {};

    for (int kk = 0; kk < I_DIM; kk += 32) {
        bf16x16 a0 = load_a_f32cvt(x + (size_t)Mbase * I_DIM + kk, I_DIM, lane);
        bf16x16 a1 = load_a_f32cvt(x + (size_t)(Mbase + 16) * I_DIM + kk, I_DIM, lane);
        bf16x16 b0 = load_b_bf16(Winb + (size_t)Nbase * I_DIM + kk, I_DIM, lane);
        bf16x16 b1 = load_b_bf16(Winb + (size_t)(Nbase + 16) * I_DIM + kk, I_DIM, lane);
        acc[0][0] = __builtin_amdgcn_wmma_f32_16x16x32_bf16(false, a0, false, b0, (short)0, acc[0][0], false, false);
        acc[0][1] = __builtin_amdgcn_wmma_f32_16x16x32_bf16(false, a0, false, b1, (short)0, acc[0][1], false, false);
        acc[1][0] = __builtin_amdgcn_wmma_f32_16x16x32_bf16(false, a1, false, b0, (short)0, acc[1][0], false, false);
        acc[1][1] = __builtin_amdgcn_wmma_f32_16x16x32_bf16(false, a1, false, b1, (short)0, acc[1][1], false, false);
    }

    int c = lane & 15, half = lane >> 4;
#pragma unroll
    for (int ni = 0; ni < 2; ++ni) {
        int n = Nbase + ni * 16 + c;
        float bias = b_in[n];
#pragma unroll
        for (int mi = 0; mi < 2; ++mi) {
#pragma unroll
            for (int r = 0; r < 8; ++r) {
                int m = Mbase + mi * 16 + half * 8 + r;   // C layout: lanes 16-31 hold M=r+8
                xproj[(size_t)m * H_DIM + n] = (__bf16)(acc[mi][ni][r] + bias);
            }
        }
    }
}

// ---------------------------------------------------------------------------
// Phase 2: sequential scan. Persistent 16 WGs; WG wg owns columns
// [wg*32, wg*32+32) of h. W_h slice (32x512 bf16 = 32KB) lives in LDS,
// staged by the Tensor Data Mover (one D# issued by wave 0).
// Each step: h_new = relu(xproj[t] + h @ W_h^T + b_h);
//            h = h*(1-a) + a*h_new;  Y[t] = h (bf16).
// Full h (bf16, 128KB) is exchanged through L2 with double buffering and a
// device-scope software barrier between steps.
__global__ __launch_bounds__(256) void rnn_scan_k(
    const __bf16* __restrict__ xproj, const __bf16* __restrict__ Whb,
    const float* __restrict__ b_h, float* __restrict__ h_f32,
    __bf16* __restrict__ hb0, __bf16* __restrict__ hb1,
    __bf16* __restrict__ Y, unsigned* __restrict__ cnt) {
    __shared__ __bf16 whs[32 * H_DIM];   // 32KB: this WG's rows of W_h (bf16)

    int wg = blockIdx.x;                 // 0..15
    int n0 = wg * 32;
    int lane = threadIdx.x & 31;
    int wave = threadIdx.x >> 5;         // 0..7 -> M tile (16 rows each, 128 = B)
    int Mbase = wave * 16;
    int c = lane & 15, half = lane >> 4;

    // Stage W_h slice (2D tile: 512 x 32 rows of 2-byte elems) into LDS via TDM.
    // D# layout per CDNA5 ISA ch.8: group0 = {flags, lds_addr, global_addr, type},
    // group1 = {data_size, tensor dims, tile dims, dim0 stride}. Groups 2/3 zero
    // (tile_dim2 = 0 -> 2D). Low 32 bits of the flat shared pointer are the LDS
    // byte address (ISA 10.2 aperture rules).
    if (threadIdx.x < 32) {
        unsigned lds_base = (unsigned)(uintptr_t)(&whs[0]);
        unsigned long long ga = (unsigned long long)(uintptr_t)(Whb + (size_t)n0 * H_DIM);
        u32x4 g0;
        g0[0] = 1u;                                             // count=1, user D#
        g0[1] = lds_base;                                       // lds_addr
        g0[2] = (unsigned)(ga & 0xFFFFFFFFu);                   // global_addr[31:0]
        g0[3] = (unsigned)((ga >> 32) & 0x01FFFFFFu) | (2u << 30); // addr[56:32] | type=2
        i32x8 g1;
        g1[0] = (int)(1u << 16);        // data_size=1 (2 bytes); workgroup_mask=0
        g1[1] = (int)((unsigned)H_DIM << 16); // tensor_dim0[15:0] in [31:16]
        g1[2] = (int)(32u << 16);       // tensor_dim0 hi=0; tensor_dim1[15:0]=32
        g1[3] = (int)((unsigned)H_DIM << 16); // tensor_dim1 hi=0; tile_dim0=512
        g1[4] = 32;                     // tile_dim1=32; tile_dim2=0 (2D)
        g1[5] = H_DIM;                  // tensor_dim0_stride[31:0]=512
        g1[6] = 0;                      // stride0 hi=0; tensor_dim1_stride lo=0
        g1[7] = 0;
        i32x4 z4 = {0, 0, 0, 0};
        i32x8 z8 = {0, 0, 0, 0, 0, 0, 0, 0};
        __builtin_amdgcn_tensor_load_to_lds(g0, g1, z4, z4, z8, 0);
        __builtin_amdgcn_s_wait_tensorcnt(0);
    }
    __syncthreads();

    float bh0 = b_h[n0 + c];
    float bh1 = b_h[n0 + 16 + c];

    for (int t = 0; t < T_DIM; ++t) {
        const __bf16* hin = (t & 1) ? hb1 : hb0;
        __bf16* hout = (t & 1) ? hb0 : hb1;

        const __bf16* xp_base = xproj + ((size_t)t * B_DIM + Mbase) * H_DIM + n0;
        __builtin_prefetch(xp_base, 0, 1);   // global_prefetch of epilogue data

        f32x8 acc0 = {}, acc1 = {};
        for (int kk = 0; kk < H_DIM; kk += 32) {
            bf16x16 a = load_a_bf16(hin + (size_t)Mbase * H_DIM + kk, H_DIM, lane);
            // B fragments straight from LDS (ds_load_b128)
            int kb = kk + half * 8;
            bf16x8 p00 = *(const bf16x8*)(&whs[(0 * 16 + c) * H_DIM + kb]);
            bf16x8 p01 = *(const bf16x8*)(&whs[(0 * 16 + c) * H_DIM + kb + 16]);
            bf16x8 p10 = *(const bf16x8*)(&whs[(1 * 16 + c) * H_DIM + kb]);
            bf16x8 p11 = *(const bf16x8*)(&whs[(1 * 16 + c) * H_DIM + kb + 16]);
            bf16x16 b0v = combine16(p00, p01);
            bf16x16 b1v = combine16(p10, p11);
            acc0 = __builtin_amdgcn_wmma_f32_16x16x32_bf16(false, a, false, b0v, (short)0, acc0, false, false);
            acc1 = __builtin_amdgcn_wmma_f32_16x16x32_bf16(false, a, false, b1v, (short)0, acc1, false, false);
        }

        // Epilogue: leaky integration + ReLU; write f32 state, bf16 state, Y
#pragma unroll
        for (int r = 0; r < 8; ++r) {
            int m = Mbase + half * 8 + r;
            size_t row = (size_t)t * B_DIM + m;
            {
                int n = n0 + c;
                float xpv = (float)xproj[row * H_DIM + n];
                float hn = acc0[r] + xpv + bh0;
                hn = hn > 0.0f ? hn : 0.0f;
                size_t idx = (size_t)m * H_DIM + n;
                float ho = h_f32[idx] * (1.0f - ALPHA_F) + ALPHA_F * hn;
                h_f32[idx] = ho;
                __bf16 hb = (__bf16)ho;
                hout[idx] = hb;
                Y[row * H_DIM + n] = hb;
            }
            {
                int n = n0 + 16 + c;
                float xpv = (float)xproj[row * H_DIM + n];
                float hn = acc1[r] + xpv + bh1;
                hn = hn > 0.0f ? hn : 0.0f;
                size_t idx = (size_t)m * H_DIM + n;
                float ho = h_f32[idx] * (1.0f - ALPHA_F) + ALPHA_F * hn;
                h_f32[idx] = ho;
                __bf16 hb = (__bf16)ho;
                hout[idx] = hb;
                Y[row * H_DIM + n] = hb;
            }
        }

        // Device-scope software barrier across the 16 persistent WGs
        __threadfence();
        __syncthreads();
        if (threadIdx.x == 0) {
            atomicAdd(cnt, 1u);
            unsigned tgt = (unsigned)(NWG_SCAN * (t + 1));
            while (__hip_atomic_load(cnt, __ATOMIC_ACQUIRE, __HIP_MEMORY_SCOPE_AGENT) < tgt)
                __builtin_amdgcn_s_sleep(8);
        }
        __syncthreads();
    }
}

// ---------------------------------------------------------------------------
// Phase 3: logits = Y @ W_out^T + b_out, softmax over O=4. One thread per row.
__global__ __launch_bounds__(256) void rnn_out_k(
    const __bf16* __restrict__ Y, const float* __restrict__ Wout,
    const float* __restrict__ bout, float* __restrict__ out) {
    __shared__ float wsm[O_DIM * H_DIM];   // 8KB
    for (int i = threadIdx.x; i < O_DIM * H_DIM; i += 256) wsm[i] = Wout[i];
    __syncthreads();

    int row = blockIdx.x * 256 + threadIdx.x;   // grid sized exactly: T*B/256
    const __bf16* y = Y + (size_t)row * H_DIM;
    float l0 = bout[0], l1 = bout[1], l2 = bout[2], l3 = bout[3];
    for (int j = 0; j < H_DIM; j += 8) {
        bf16x8 v = *(const bf16x8*)(y + j);
#pragma unroll
        for (int e = 0; e < 8; ++e) {
            float f = (float)v[e];
            l0 += f * wsm[0 * H_DIM + j + e];
            l1 += f * wsm[1 * H_DIM + j + e];
            l2 += f * wsm[2 * H_DIM + j + e];
            l3 += f * wsm[3 * H_DIM + j + e];
        }
    }
    float mx = fmaxf(fmaxf(l0, l1), fmaxf(l2, l3));
    float e0 = __expf(l0 - mx), e1 = __expf(l1 - mx), e2 = __expf(l2 - mx), e3 = __expf(l3 - mx);
    float inv = 1.0f / (e0 + e1 + e2 + e3);
    float* o = out + (size_t)row * O_DIM;
    o[0] = e0 * inv; o[1] = e1 * inv; o[2] = e2 * inv; o[3] = e3 * inv;
}

// Copy final f32 state into the tail of d_out
__global__ __launch_bounds__(256) void rnn_state_k(
    const float* __restrict__ h, float* __restrict__ out_tail) {
    int i = blockIdx.x * 256 + threadIdx.x;
    if (i < B_DIM * H_DIM) out_tail[i] = h[i];
}

// ---------------------------------------------------------------------------
extern "C" void kernel_launch(void* const* d_in, const int* in_sizes, int n_in,
                              void* d_out, int out_size, void* d_ws, size_t ws_size,
                              hipStream_t stream) {
    (void)in_sizes; (void)n_in; (void)out_size; (void)ws_size;
    const float* x     = (const float*)d_in[0];
    const float* W_in  = (const float*)d_in[1];
    const float* b_in  = (const float*)d_in[2];
    const float* W_h   = (const float*)d_in[3];
    const float* b_h   = (const float*)d_in[4];
    const float* W_out = (const float*)d_in[5];
    const float* b_out = (const float*)d_in[6];
    float* out = (float*)d_out;

    // Workspace carve-up (256B aligned). Total ~270 MB.
    char* ws = (char*)d_ws;
    size_t off = 0;
    auto carve = [&](size_t bytes) -> void* {
        void* p = ws + off;
        off += bytes;
        off = (off + 255) & ~(size_t)255;
        return p;
    };
    unsigned* cnt   = (unsigned*)carve(sizeof(unsigned));
    float*    h_f32 = (float*)carve((size_t)B_DIM * H_DIM * sizeof(float));
    __bf16*   hb0   = (__bf16*)carve((size_t)B_DIM * H_DIM * sizeof(__bf16));
    __bf16*   hb1   = (__bf16*)carve((size_t)B_DIM * H_DIM * sizeof(__bf16));
    __bf16*   Winb  = (__bf16*)carve((size_t)H_DIM * I_DIM * sizeof(__bf16));
    __bf16*   Whb   = (__bf16*)carve((size_t)H_DIM * H_DIM * sizeof(__bf16));
    __bf16*   xproj = (__bf16*)carve((size_t)T_DIM * B_DIM * H_DIM * sizeof(__bf16));
    __bf16*   Ybuf  = (__bf16*)carve((size_t)T_DIM * B_DIM * H_DIM * sizeof(__bf16));

    rnn_init_k<<<(B_DIM * H_DIM + 255) / 256, 256, 0, stream>>>(h_f32, hb0, hb1, cnt);
    rnn_cvt_weights_k<<<(H_DIM * I_DIM + 255) / 256, 256, 0, stream>>>(W_in, W_h, Winb, Whb);

    dim3 gGemm(H_DIM / 64, (T_DIM * B_DIM) / 128);   // (8, 1024)
    rnn_xproj_gemm_k<<<gGemm, 256, 0, stream>>>(x, Winb, b_in, xproj);

    rnn_scan_k<<<NWG_SCAN, 256, 0, stream>>>(xproj, Whb, b_h, h_f32, hb0, hb1, Ybuf, cnt);

    rnn_out_k<<<(T_DIM * B_DIM) / 256, 256, 0, stream>>>(Ybuf, W_out, b_out, out);
    rnn_state_k<<<(B_DIM * H_DIM + 255) / 256, 256, 0, stream>>>(
        h_f32, out + (size_t)T_DIM * B_DIM * O_DIM);
}
